// E2EModelProved_89180700934288
// MI455X (gfx1250) — compile-verified, matrix operand order checked by reading, not run on hardware
//
#include <hip/hip_runtime.h>

// ---------------------------------------------------------------------------
// Model dims
// ---------------------------------------------------------------------------
#define SRC_LEN 64
#define DEC_LEN 48
#define BATCH   32
#define EMB     512
#define HID     1024     // ENC_H == DEC_H == A_DIM
#define G3      3072     // 3*HID
#define VOCAB   32000
#define BOS_ID  1

typedef float  v8f   __attribute__((ext_vector_type(8)));
typedef __bf16 v16bf __attribute__((ext_vector_type(16)));
typedef __bf16 bf16x8 __attribute__((ext_vector_type(8)));

// f32 -> bf16 round-to-nearest-even
static __device__ __forceinline__ __bf16 f2bf(float f) {
    unsigned u = __builtin_bit_cast(unsigned, f);
    u += 0x7FFFu + ((u >> 16) & 1u);
    unsigned short s = (unsigned short)(u >> 16);
    return __builtin_bit_cast(__bf16, s);
}

// Load a 16-element bf16 WMMA fragment for this lane.
// Per CDNA5 ISA 16-bit A/B layout: lane (r, half) holds K = base+[0..7] in
// elements 0..7 and K = base+16+[0..7] in elements 8..15, base = half*8.
// Caller passes ptr = row_base + half*8 (+ k0).
static __device__ __forceinline__ v16bf load_frag(const __bf16* p) {
    union { v16bf v; bf16x8 h[2]; } u;
    u.h[0] = *(const bf16x8*)(p);
    u.h[1] = *(const bf16x8*)(p + 16);
    return u.v;
}

// ---------------------------------------------------------------------------
// Generic NT GEMM:  C[M,N] = A[M,K] * W[N,K]^T (+ bias[N]) ; C f32, optional
// bf16 mirror Cbf for chained GEMMs. One wave per block; each wave computes a
// 32(M) x 64(N) tile via v_wmma_f32_16x16x32_bf16: 2 A fragments x 4 B
// fragments -> 8 WMMAs per K-step of 32, 2 b128 loads per fragment.
// Requires: M%32==0, N%64==0, K%32==0 (true for all uses here).
// ---------------------------------------------------------------------------
__global__ void gemm_nt_bf16_kernel(const __bf16* __restrict__ A, int lda,
                                    const __bf16* __restrict__ W, int ldb,
                                    float* __restrict__ C, int ldc,
                                    const float* __restrict__ bias,
                                    __bf16* __restrict__ Cbf, int K) {
    const int n0   = blockIdx.x * 64;
    const int m0   = blockIdx.y * 32;
    const int lane = threadIdx.x & 31;
    const int r    = lane & 15;
    const int half = lane >> 4;

    v8f acc[2][4] = {};
    const __bf16* ap0 = A + (size_t)(m0 + r) * lda + half * 8;
    const __bf16* ap1 = ap0 + (size_t)16 * lda;
    const __bf16* wp  = W + (size_t)(n0 + r) * ldb + half * 8;

    for (int k = 0; k < K; k += 32) {
        v16bf a0 = load_frag(ap0 + k);
        v16bf a1 = load_frag(ap1 + k);
#pragma unroll
        for (int j = 0; j < 4; ++j) {
            v16bf b = load_frag(wp + (size_t)(16 * j) * ldb + k);
            acc[0][j] = __builtin_amdgcn_wmma_f32_16x16x32_bf16(
                false, a0, false, b, (short)0, acc[0][j], false, false);
            acc[1][j] = __builtin_amdgcn_wmma_f32_16x16x32_bf16(
                false, a1, false, b, (short)0, acc[1][j], false, false);
        }
    }

    // C/D layout: VGPR v, lanes 0-15 -> (M=v, N=lane); lanes 16-31 -> (M=v+8)
#pragma unroll
    for (int j = 0; j < 4; ++j) {
        const int n  = n0 + 16 * j + r;
        const float bv = bias ? bias[n] : 0.0f;
#pragma unroll
        for (int mt = 0; mt < 2; ++mt) {
#pragma unroll
            for (int v = 0; v < 8; ++v) {
                const int m = m0 + 16 * mt + v + 8 * half;
                const float x = acc[mt][j][v] + bv;
                C[(size_t)m * ldc + n] = x;
                if (Cbf) Cbf[(size_t)m * ldc + n] = f2bf(x);
            }
        }
    }
}

// ---------------------------------------------------------------------------
// Small utility kernels
// ---------------------------------------------------------------------------
__global__ void f32_to_bf16_kernel(const float* __restrict__ src,
                                   __bf16* __restrict__ dst, int n) {
    for (int i = blockIdx.x * blockDim.x + threadIdx.x; i < n;
         i += gridDim.x * blockDim.x)
        dst[i] = f2bf(src[i]);
}

__global__ void zero_u32_kernel(unsigned* __restrict__ p, int n) {
    for (int i = blockIdx.x * blockDim.x + threadIdx.x; i < n;
         i += gridDim.x * blockDim.x)
        p[i] = 0u;
}

// Encoder input embeddings: one block per (s,b) token.
__global__ void gather_enc_kernel(const int* __restrict__ ids,
                                  const float* __restrict__ emb,
                                  __bf16* __restrict__ dst) {
    const int p  = blockIdx.x;
    const int id = ids[p];
    const float* s = emb + (size_t)id * EMB;
    __bf16* d     = dst + (size_t)p * EMB;
    for (int j = threadIdx.x; j < EMB; j += blockDim.x) d[j] = f2bf(s[j]);
}

// Teacher-forced decoder inputs: BOS at t=0, else batch_y[t-1].
__global__ void gather_dec_kernel(const int* __restrict__ batch_y,
                                  const float* __restrict__ emb,
                                  __bf16* __restrict__ dst) {
    const int p = blockIdx.x;
    const int t = p >> 5, b = p & 31;
    const int id = (t == 0) ? BOS_ID : batch_y[(t - 1) * BATCH + b];
    const float* s = emb + (size_t)id * EMB;
    __bf16* d     = dst + (size_t)p * EMB;
    for (int j = threadIdx.x; j < EMB; j += blockDim.x) d[j] = f2bf(s[j]);
}

// GRU gate fusion (PyTorch GRUCell math). 32x1024 elements.
__global__ void gru_gates_kernel(const float* __restrict__ gi,
                                 const float* __restrict__ gh,
                                 const float* __restrict__ h,
                                 float* __restrict__ hnew,
                                 __bf16* __restrict__ hnew_bf,
                                 float* __restrict__ extra_f,
                                 __bf16* __restrict__ extra_bf) {
    const int idx = blockIdx.x * blockDim.x + threadIdx.x;   // 0..32767
    const int b = idx >> 10, j = idx & 1023;
    const float* gib = gi + (size_t)b * G3;
    const float* ghb = gh + (size_t)b * G3;
    const float ir = gib[j], iz = gib[HID + j], in = gib[2 * HID + j];
    const float hr = ghb[j], hz = ghb[HID + j], hn = ghb[2 * HID + j];
    const float rg = 1.0f / (1.0f + expf(-(ir + hr)));
    const float zg = 1.0f / (1.0f + expf(-(iz + hz)));
    const float ng = tanhf(in + rg * hn);
    const float h2 = (1.0f - zg) * ng + zg * h[idx];
    hnew[idx]    = h2;
    hnew_bf[idx] = f2bf(h2);
    if (extra_f)  extra_f[idx]  = h2;
    if (extra_bf) extra_bf[idx] = f2bf(h2);
}

// Attention for one decoder step t: scores (with the reference's faithful
// view(S*B)->view(B,S) index permutation and multiplicative mask), softmax,
// context, and assembly of rnn_in = [prev_y | ctx] as bf16.
// One block per batch element b.
__global__ void attn_kernel(const float* __restrict__ uh,      // (S*B, HID)
                            const float* __restrict__ wq,      // (B, HID)
                            const float* __restrict__ v_w,     // (HID)
                            const float* __restrict__ v_b,     // (1)
                            const float* __restrict__ enc_out, // (S*B, HID)
                            const __bf16* __restrict__ Ydec,   // (T*B, EMB)
                            __bf16* __restrict__ rnnin,        // (B, EMB+HID)
                            int t) {
    const int b    = blockIdx.x;
    const int tid  = threadIdx.x;
    const int lane = tid & 31, wave = tid >> 5;
    __shared__ float sc[SRC_LEN];

    // scores[b][s_] = v . tanh(uh[su,bu,:] + wq[bu,:]) + v_b,
    // where flat i = b*S + s_, su = i/B, bu = i%B  (the reshape quirk).
    for (int s_ = wave; s_ < SRC_LEN; s_ += 8) {
        const int i  = b * SRC_LEN + s_;
        const int su = i >> 5, bu = i & 31;
        const float* uhr = uh + ((size_t)su * BATCH + bu) * HID;
        const float* wqr = wq + (size_t)bu * HID;
        float p = 0.0f;
        for (int a = lane; a < HID; a += 32)
            p += v_w[a] * tanhf(uhr[a] + wqr[a]);
#pragma unroll
        for (int m = 16; m > 0; m >>= 1) p += __shfl_xor(p, m, 32);
        if (lane == 0) sc[s_] = p + v_b[0];
    }
    __syncthreads();

    // multiplicative mask + softmax over the 64 entries (tiny; one thread)
    if (tid == 0) {
        const int mc = (t < SRC_LEN) ? t : SRC_LEN;
        float mx = -1e30f;
        for (int s_ = 0; s_ < SRC_LEN; ++s_) {
            const float v = (s_ < mc) ? sc[s_] : 0.0f;
            sc[s_] = v;
            mx = fmaxf(mx, v);
        }
        float sum = 0.0f;
        for (int s_ = 0; s_ < SRC_LEN; ++s_) {
            const float e = expf(sc[s_] - mx);
            sc[s_] = e;
            sum += e;
        }
        const float inv = 1.0f / sum;
        for (int s_ = 0; s_ < SRC_LEN; ++s_) sc[s_] *= inv;
    }
    __syncthreads();

    // ctx[b,e] = sum_s attn[s] * enc_out[s,b,e]  -> rnnin[b, EMB + e]
    for (int e = tid; e < HID; e += blockDim.x) {
        float acc = 0.0f;
        for (int s_ = 0; s_ < SRC_LEN; ++s_)
            acc += sc[s_] * enc_out[((size_t)s_ * BATCH + b) * HID + e];
        rnnin[(size_t)b * (EMB + HID) + EMB + e] = f2bf(acc);
    }
    // prev_y -> rnnin[b, 0:EMB]
    for (int j = tid; j < EMB; j += blockDim.x)
        rnnin[(size_t)b * (EMB + HID) + j] =
            Ydec[((size_t)t * BATCH + b) * EMB + j];
}

// Row-wise log_softmax over VOCAB, in place.
__global__ void log_softmax_kernel(float* __restrict__ logits) {
    const int row = blockIdx.x;
    float* p = logits + (size_t)row * VOCAB;
    __shared__ float red[256];
    const int tid = threadIdx.x;

    float mx = -1e30f;
    for (int j = tid; j < VOCAB; j += 256) mx = fmaxf(mx, p[j]);
    red[tid] = mx; __syncthreads();
    for (int s = 128; s > 0; s >>= 1) {
        if (tid < s) red[tid] = fmaxf(red[tid], red[tid + s]);
        __syncthreads();
    }
    mx = red[0]; __syncthreads();

    float sum = 0.0f;
    for (int j = tid; j < VOCAB; j += 256) sum += expf(p[j] - mx);
    red[tid] = sum; __syncthreads();
    for (int s = 128; s > 0; s >>= 1) {
        if (tid < s) red[tid] += red[tid + s];
        __syncthreads();
    }
    const float lse = mx + logf(red[0]);
    __syncthreads();

    for (int j = tid; j < VOCAB; j += 256) p[j] = p[j] - lse;
}

// ---------------------------------------------------------------------------
// Host orchestration
// ---------------------------------------------------------------------------
static inline void launch_gemm(const __bf16* A, int lda, const __bf16* W, int ldb,
                               float* C, int ldc, const float* bias, __bf16* Cbf,
                               int M, int N, int K, hipStream_t s) {
    dim3 grid(N / 64, M / 32);
    gemm_nt_bf16_kernel<<<grid, dim3(32), 0, s>>>(A, lda, W, ldb, C, ldc, bias, Cbf, K);
}

extern "C" void kernel_launch(void* const* d_in, const int* in_sizes, int n_in,
                              void* d_out, int out_size, void* d_ws, size_t ws_size,
                              hipStream_t stream) {
    (void)in_sizes; (void)n_in; (void)out_size; (void)ws_size;

    const int*   batch_x = (const int*)  d_in[0];
    const int*   batch_y = (const int*)  d_in[1];
    const float* emb     = (const float*)d_in[2];
    const float* enc_Wih = (const float*)d_in[3];
    const float* enc_Whh = (const float*)d_in[4];
    const float* enc_bih = (const float*)d_in[5];
    const float* enc_bhh = (const float*)d_in[6];
    const float* dec_Wih = (const float*)d_in[7];
    const float* dec_Whh = (const float*)d_in[8];
    const float* dec_bih = (const float*)d_in[9];
    const float* dec_bhh = (const float*)d_in[10];
    const float* U_w     = (const float*)d_in[11];
    const float* U_b     = (const float*)d_in[12];
    const float* Wq_w    = (const float*)d_in[13];
    const float* Wq_b    = (const float*)d_in[14];
    const float* v_w     = (const float*)d_in[15];
    const float* v_b     = (const float*)d_in[16];
    const float* Wc_w    = (const float*)d_in[17];
    const float* Wc_b    = (const float*)d_in[18];
    const float* Wout_w  = (const float*)d_in[19];
    const float* Wout_b  = (const float*)d_in[20];
    float* out = (float*)d_out;

    // ---- workspace carving (~143 MB) ----
    char* ws = (char*)d_ws;
    size_t off = 0;
    auto carve = [&](size_t bytes) -> char* {
        char* p = ws + off;
        off = (off + bytes + 255) & ~(size_t)255;
        return p;
    };
    __bf16* enc_Wih_bf = (__bf16*)carve((size_t)G3 * EMB * 2);
    __bf16* enc_Whh_bf = (__bf16*)carve((size_t)G3 * HID * 2);
    __bf16* dec_Wih_bf = (__bf16*)carve((size_t)G3 * HID * 2);
    __bf16* dec_Whh_bf = (__bf16*)carve((size_t)G3 * HID * 2);
    __bf16* U_bf       = (__bf16*)carve((size_t)HID * HID * 2);
    __bf16* Wq_bf      = (__bf16*)carve((size_t)HID * HID * 2);
    __bf16* Wc_bf      = (__bf16*)carve((size_t)HID * (EMB + HID) * 2);
    __bf16* Wout_bf    = (__bf16*)carve((size_t)VOCAB * HID * 2);
    __bf16* Xemb_bf    = (__bf16*)carve((size_t)SRC_LEN * BATCH * EMB * 2);
    __bf16* Ydec_bf    = (__bf16*)carve((size_t)DEC_LEN * BATCH * EMB * 2);
    float*  gi_all     = (float*) carve((size_t)SRC_LEN * BATCH * G3 * 4);
    float*  enc_out    = (float*) carve((size_t)SRC_LEN * BATCH * HID * 4);
    __bf16* enc_out_bf = (__bf16*)carve((size_t)SRC_LEN * BATCH * HID * 2);
    float*  uh         = (float*) carve((size_t)SRC_LEN * BATCH * HID * 4);
    float*  h          = (float*) carve((size_t)BATCH * HID * 4);
    __bf16* h_bf       = (__bf16*)carve((size_t)BATCH * HID * 2);
    float*  gh         = (float*) carve((size_t)BATCH * G3 * 4);
    float*  wq         = (float*) carve((size_t)BATCH * HID * 4);
    __bf16* rnnin_bf   = (__bf16*)carve((size_t)BATCH * (EMB + HID) * 2);
    float*  rnnin2     = (float*) carve((size_t)BATCH * HID * 4);
    __bf16* rnnin2_bf  = (__bf16*)carve((size_t)BATCH * HID * 2);
    float*  gi_dec     = (float*) carve((size_t)BATCH * G3 * 4);
    __bf16* Hall_bf    = (__bf16*)carve((size_t)DEC_LEN * BATCH * HID * 2);

    // ---- one-time weight conversion f32 -> bf16 ----
    auto conv = [&](const float* s, __bf16* d, size_t n) {
        int blocks = (int)((n + 255) / 256);
        if (blocks > 4096) blocks = 4096;
        f32_to_bf16_kernel<<<blocks, 256, 0, stream>>>(s, d, (int)n);
    };
    conv(enc_Wih, enc_Wih_bf, (size_t)G3 * EMB);
    conv(enc_Whh, enc_Whh_bf, (size_t)G3 * HID);
    conv(dec_Wih, dec_Wih_bf, (size_t)G3 * HID);
    conv(dec_Whh, dec_Whh_bf, (size_t)G3 * HID);
    conv(U_w,     U_bf,       (size_t)HID * HID);
    conv(Wq_w,    Wq_bf,      (size_t)HID * HID);
    conv(Wc_w,    Wc_bf,      (size_t)HID * (EMB + HID));
    conv(Wout_w,  Wout_bf,    (size_t)VOCAB * HID);

    // ---- embedding gathers ----
    gather_enc_kernel<<<SRC_LEN * BATCH, 128, 0, stream>>>(batch_x, emb, Xemb_bf);
    gather_dec_kernel<<<DEC_LEN * BATCH, 128, 0, stream>>>(batch_y, emb, Ydec_bf);

    // ---- h0 = 0 ----
    zero_u32_kernel<<<128, 256, 0, stream>>>((unsigned*)h, BATCH * HID);
    zero_u32_kernel<<<64, 256, 0, stream>>>((unsigned*)h_bf, BATCH * HID / 2);

    // ---- encoder: batched input gates, then 64 recurrent steps ----
    launch_gemm(Xemb_bf, EMB, enc_Wih_bf, EMB, gi_all, G3, enc_bih, nullptr,
                SRC_LEN * BATCH, G3, EMB, stream);
    for (int s = 0; s < SRC_LEN; ++s) {
        launch_gemm(h_bf, HID, enc_Whh_bf, HID, gh, G3, enc_bhh, nullptr,
                    BATCH, G3, HID, stream);
        gru_gates_kernel<<<128, 256, 0, stream>>>(
            gi_all + (size_t)s * BATCH * G3, gh, h, h, h_bf,
            enc_out + (size_t)s * BATCH * HID,
            enc_out_bf + (size_t)s * BATCH * HID);
    }

    // ---- step-invariant attention projection uh = enc_out @ U^T + U_b ----
    launch_gemm(enc_out_bf, HID, U_bf, HID, uh, HID, U_b, nullptr,
                SRC_LEN * BATCH, HID, HID, stream);

    // ---- decoder: 48 recurrent steps (h carries over from encoder) ----
    for (int t = 0; t < DEC_LEN; ++t) {
        launch_gemm(h_bf, HID, Wq_bf, HID, wq, HID, Wq_b, nullptr,
                    BATCH, HID, HID, stream);
        attn_kernel<<<BATCH, 256, 0, stream>>>(uh, wq, v_w, v_b, enc_out,
                                               Ydec_bf, rnnin_bf, t);
        launch_gemm(rnnin_bf, EMB + HID, Wc_bf, EMB + HID, rnnin2, HID, Wc_b,
                    rnnin2_bf, BATCH, HID, EMB + HID, stream);
        launch_gemm(rnnin2_bf, HID, dec_Wih_bf, HID, gi_dec, G3, dec_bih,
                    nullptr, BATCH, G3, HID, stream);
        launch_gemm(h_bf, HID, dec_Whh_bf, HID, gh, G3, dec_bhh, nullptr,
                    BATCH, G3, HID, stream);
        gru_gates_kernel<<<128, 256, 0, stream>>>(
            gi_dec, gh, h, h, h_bf, nullptr,
            Hall_bf + (size_t)t * BATCH * HID);
    }

    // ---- deferred output projection (the 100-GFLOP GEMM) + log_softmax ----
    launch_gemm(Hall_bf, HID, Wout_bf, HID, out, VOCAB, Wout_b, nullptr,
                DEC_LEN * BATCH, VOCAB, HID, stream);
    log_softmax_kernel<<<DEC_LEN * BATCH, 256, 0, stream>>>(out);
}
